// SparseConv3DBlock_3058016715333
// MI455X (gfx1250) — matrix operand order repaired and checked by reading, not compiled, not run on hardware
//
#include <hip/hip_runtime.h>

typedef __attribute__((ext_vector_type(16))) _Float16 v16h;
typedef __attribute__((ext_vector_type(8)))  _Float16 v8h;
typedef __attribute__((ext_vector_type(4)))  _Float16 v4h;
typedef __attribute__((ext_vector_type(8)))  float    v8f;
typedef __attribute__((ext_vector_type(4)))  float    v4f;

#define NVOX 200000
#define FIN  64
#define FOUT 128
#define KOFF 27
#define MTILE 64          // voxels per block (4 WMMA row tiles)

// ---------------------------------------------------------------------------
// Async global->LDS copy of 16 bytes per lane (ASYNCcnt-tracked).
// ---------------------------------------------------------------------------
__device__ __forceinline__ void async_copy16(unsigned lds_off, const void* g)
{
    unsigned long long ga = (unsigned long long)g;
    asm volatile("global_load_async_to_lds_b128 %0, %1, off"
                 :: "v"(lds_off), "v"(ga) : "memory");
}

__device__ __forceinline__ void wait_async0()
{
#if __has_builtin(__builtin_amdgcn_s_wait_asynccnt)
    __builtin_amdgcn_s_wait_asynccnt(0);
#else
    asm volatile("s_wait_asynccnt 0x0" ::: "memory");
#endif
}

// ---------------------------------------------------------------------------
// Kernel 1: fused BatchNorm(inference) + SiLU, emit f16 activation table.
// ---------------------------------------------------------------------------
__global__ __launch_bounds__(256) void bn_silu_f16(
    const float* __restrict__ x,
    const float* __restrict__ gamma, const float* __restrict__ beta,
    const float* __restrict__ rmean, const float* __restrict__ rvar,
    _Float16* __restrict__ yh, int total4)
{
    int t = blockIdx.x * blockDim.x + threadIdx.x;
    if (t >= total4) return;
    int e = t * 4;
    int f = e & (FIN - 1);
    v4f xv = *(const v4f*)(x + e);
    v4h o;
#pragma unroll
    for (int i = 0; i < 4; ++i) {
        float s = gamma[f + i] * rsqrtf(rvar[f + i] + 1e-5f);
        float v = (xv[i] - rmean[f + i]) * s + beta[f + i];
        o[i] = (_Float16)(v / (1.0f + __expf(-v)));   // silu
    }
    *(v4h*)(yh + e) = o;
}

// ---------------------------------------------------------------------------
// Kernel 2: pack W [27][64][128] f32 -> B-operand f16 tiles.
// Tile (k, s, ct): 32(K) x 16(N), stored [n][kk]; lane reads 32 contiguous B.
// ---------------------------------------------------------------------------
__global__ __launch_bounds__(256) void wt_pack(
    const float* __restrict__ W, _Float16* __restrict__ Wt, int total)
{
    int e = blockIdx.x * blockDim.x + threadIdx.x;
    if (e >= total) return;
    int kk = e & 31;
    int n  = (e >> 5) & 15;
    int ct = (e >> 9) & 7;
    int s  = (e >> 12) & 1;
    int k  = e >> 13;
    Wt[e] = (_Float16)W[(size_t)(k * 64 + s * 32 + kk) * FOUT + ct * 16 + n];
}

// ---------------------------------------------------------------------------
// Kernel 3: implicit GEMM. Block = 256 thr = 8 waves; block owns 64 voxels,
// wave ct owns 16 output cols with 4 accumulators (row tiles).
//
// LDS layout is WMMA-fragment order: region g = s*2+half (4 x 2 KB), slot =
// 32 B per row. A lane's A-operand fragment for (rt, s) is one contiguous,
// 32 B-aligned chunk -> single v16h ds read, no register shuffles. The async
// gather performs the permutation for free (two b128 per thread with
// different global offsets).
// ---------------------------------------------------------------------------
__global__ __launch_bounds__(256) void spconv_wmma(
    const _Float16* __restrict__ yh,
    const _Float16* __restrict__ Wt,
    const int* __restrict__ nidx,
    float* __restrict__ out)
{
    __shared__ __align__(64) _Float16 sA[2][4 * 1024];   // 2 x 8 KB

    const int tid  = threadIdx.x;
    const int lane = tid & 31;
    const int ct   = tid >> 5;          // wave id == column tile 0..7
    const int half = lane >> 4;
    const int mr   = lane & 15;
    const int row0 = blockIdx.x * MTILE;

    // staging role: thread t gathers fragment-group g=(t&3) of row (t>>2)
    const int srow = tid >> 2;
    const int sg   = tid & 3;           // g = s*2 + half
    const int ss   = sg >> 1;
    const int sh   = sg & 1;
    int grow = row0 + srow;
    if (grow >= NVOX) grow = NVOX - 1;
    const int* idxp = nidx + (size_t)grow * KOFF;

    // byte offsets within a gathered 128 B activation row
    const int glo = ss * 64 + sh * 32;        // K = s*32 + half*16 + 0..7
    const int ghi = glo + 16;                 // K = s*32 + half*16 + 8..15
    const unsigned lfrag = sg * 2048 + srow * 32;   // LDS byte offset in buf

    v8f acc[4] = {};

    // prologue: stage k = 0 into buffer 0
    {
        const char* g = (const char*)(yh + (size_t)idxp[0] * FIN);
        unsigned l = (unsigned)(uintptr_t)&sA[0][0] + lfrag;
        async_copy16(l,      g + glo);
        async_copy16(l + 16, g + ghi);
    }
    wait_async0();
    __syncthreads();

    for (int k = 0; k < KOFF; ++k) {
        const int buf = k & 1;
        if (k < KOFF - 1) {            // stage k+1 into the other buffer
            const char* g = (const char*)(yh + (size_t)idxp[k + 1] * FIN);
            unsigned l = (unsigned)(uintptr_t)&sA[buf ^ 1][0] + lfrag;
            async_copy16(l,      g + glo);
            async_copy16(l + 16, g + ghi);
        }

        const _Float16* As = &sA[buf][0];
#pragma unroll
        for (int s = 0; s < 2; ++s) {
            // B: lane holds col mr, K = half*16 + 0..15 (32 contiguous bytes)
            const _Float16* bt = Wt + ((size_t)(k * 2 + s) * 8 + ct) * 512
                                    + mr * 32 + half * 16;
            v16h b = *(const v16h*)bt;
            const _Float16* Ag = As + (s * 2 + half) * 1024;
#pragma unroll
            for (int rt = 0; rt < 4; ++rt) {
                v16h a = *(const v16h*)(Ag + (rt * 16 + mr) * 16);
                acc[rt] = __builtin_amdgcn_wmma_f32_16x16x32_f16(
                              false, a, false, b, (short)0, acc[rt],
                              false, false);
            }
        }

        if (k < KOFF - 1) wait_async0();   // my async stores to LDS landed
        __syncthreads();                   // everyone's portion visible; buf
                                           // k&1 fully consumed before reuse
    }

    // C/D f32 layout: lane, VGPR v -> row v + 8*half, col mr
    const int col = ct * 16 + mr;
#pragma unroll
    for (int rt = 0; rt < 4; ++rt) {
#pragma unroll
        for (int v = 0; v < 8; ++v) {
            int m = row0 + rt * 16 + v + half * 8;
            if (m < NVOX) out[(size_t)m * FOUT + col] = acc[rt][v];
        }
    }
}

// ---------------------------------------------------------------------------
extern "C" void kernel_launch(void* const* d_in, const int* in_sizes, int n_in,
                              void* d_out, int out_size, void* d_ws, size_t ws_size,
                              hipStream_t stream)
{
    const float* x     = (const float*)d_in[0];
    const float* gamma = (const float*)d_in[1];
    const float* beta  = (const float*)d_in[2];
    const float* rmean = (const float*)d_in[3];
    const float* rvar  = (const float*)d_in[4];
    const float* W     = (const float*)d_in[5];
    const int*   nidx  = (const int*)d_in[6];
    float*       out   = (float*)d_out;

    _Float16* Wt = (_Float16*)d_ws;                        // 442,368 B
    _Float16* yh = (_Float16*)((char*)d_ws + (1u << 19));  // 25.6 MB

    int total4 = NVOX * FIN / 4;
    bn_silu_f16<<<(total4 + 255) / 256, 256, 0, stream>>>(
        x, gamma, beta, rmean, rvar, yh, total4);

    int wtot = KOFF * 2 * 8 * 512;
    wt_pack<<<(wtot + 255) / 256, 256, 0, stream>>>(W, Wt, wtot);

    spconv_wmma<<<(NVOX + MTILE - 1) / MTILE, 256, 0, stream>>>(yh, Wt, nidx, out);
}
